// TextEncoder_30356828848486
// MI455X (gfx1250) — matrix operand order repaired and checked by reading, not compiled, not run on hardware
//
#include <hip/hip_runtime.h>
#include <hip/hip_bf16.h>
#include <stdint.h>

#define B_ROWS 4096
#define D_IN   768
#define D_OUT  1024
#define N_EXP  10

#define TM 64
#define TN 128
#define TK 32
#define LDAP 40          // padded LDS row length in bf16 elems (80B, 16B-aligned, conflict-free)
#define NTHREADS 256

typedef __attribute__((ext_vector_type(16))) __bf16 bf16x16;
typedef __attribute__((ext_vector_type(8)))  float  f32x8;

union FragAB { bf16x16 v; uint4 q[2]; };

// ---- f32 -> bf16 (RNE), scalar and packed-pair forms ----

__device__ __forceinline__ unsigned short f2bf(float f) {
  unsigned int u = __float_as_uint(f);
  unsigned int r = u + 0x7FFFu + ((u >> 16) & 1u);
  return (unsigned short)(r >> 16);
}

// returns {lo -> [15:0], hi -> [31:16]}
__device__ __forceinline__ unsigned int f2bf2(float lo, float hi) {
#if defined(__gfx1250__) && __has_builtin(__builtin_amdgcn_cvt_pk_bf16_f32)
  typedef __attribute__((ext_vector_type(2))) __bf16 bf16x2;
  union { bf16x2 v; unsigned int u; } cv;
  cv.v = __builtin_amdgcn_cvt_pk_bf16_f32(lo, hi);
  return cv.u;
#else
  unsigned int a = __float_as_uint(lo); a += 0x7FFFu + ((a >> 16) & 1u);
  unsigned int b = __float_as_uint(hi); b += 0x7FFFu + ((b >> 16) & 1u);
  return (a >> 16) | (b & 0xFFFF0000u);
#endif
}

__device__ __forceinline__ void wait_asynccnt0() {
#if __has_builtin(__builtin_amdgcn_s_wait_asynccnt)
  __builtin_amdgcn_s_wait_asynccnt(0);
#else
  asm volatile("s_wait_asynccnt 0" ::: "memory");
#endif
}

// ---------------- routing ----------------

__global__ void init_counts_kernel(int* counts, int* cursor) {
  int t = threadIdx.x;
  if (t < N_EXP) { counts[t] = 0; cursor[t] = 0; }
}

__global__ void route_kernel(const float* __restrict__ logits, int* __restrict__ cls,
                             int* __restrict__ counts, float* __restrict__ outLogits) {
  int b = blockIdx.x * blockDim.x + threadIdx.x;
  if (b >= B_ROWS) return;
  const float* row = logits + (size_t)b * N_EXP;
  float best = row[0]; int bi = 0;
  outLogits[(size_t)b * N_EXP] = best;
  for (int j = 1; j < N_EXP; ++j) {
    float v = row[j];
    outLogits[(size_t)b * N_EXP + j] = v;
    if (v > best) { best = v; bi = j; }          // first-max tie break, matches jnp.argmax
  }
  cls[b] = bi;
  atomicAdd(&counts[bi], 1);
}

__global__ void scan_kernel(const int* __restrict__ counts, int* __restrict__ segOff) {
  if (threadIdx.x == 0) {
    int acc = 0;
    segOff[0] = 0;
    for (int e = 0; e < N_EXP; ++e) { acc += counts[e]; segOff[e + 1] = acc; }
  }
}

__global__ void scatter_kernel(const int* __restrict__ cls, const int* __restrict__ segOff,
                               int* __restrict__ cursor, int* __restrict__ perm) {
  int b = blockIdx.x * blockDim.x + threadIdx.x;
  if (b >= B_ROWS) return;
  int e = cls[b];
  int p = segOff[e] + atomicAdd(&cursor[e], 1);
  perm[p] = b;
}

// ---------------- grouped GEMM (bf16 WMMA, f32 accumulate) ----------------
// PHASE 1: embed1 = x[perm] @ W1[e];  also h = bf16(gelu(embed1))
// PHASE 2: s = embed1 + h @ W2[e];    s overwrites embed1 buffer in place

template <int PHASE>
__global__ __launch_bounds__(NTHREADS)
void moe_gemm_kernel(const float* __restrict__ xf,           // phase1 A source (f32, gathered)
                     const unsigned short* __restrict__ hA,  // phase2 A source (bf16, contiguous)
                     const float* __restrict__ W,            // expert weights [E, K, 1024] f32
                     const int* __restrict__ perm,
                     const int* __restrict__ segOff,
                     float* __restrict__ eBuf,               // embed1 (phase1 out / phase2 in+out)
                     unsigned short* __restrict__ hOut,      // phase1 gelu output (bf16)
                     int K) {
  __shared__ unsigned short ldsA[TM * LDAP];
  __shared__ unsigned short ldsB[TN * LDAP];
  __shared__ int rowIdx[TM];

  const int e = blockIdx.z;
  const int segS = segOff[e], segE = segOff[e + 1];
  const int posBase = segS + blockIdx.y * TM;
  if (posBase >= segE) return;
  const int nBase = blockIdx.x * TN;
  const int tid = threadIdx.x;

  if (PHASE == 1 && tid < TM) {
    int p = posBase + tid;
    rowIdx[tid] = (p < segE) ? perm[p] : -1;
  }
  __syncthreads();

  const float* Wb = W + (size_t)e * (size_t)K * D_OUT;
  const int lane = tid & 31;
  const int wave = tid >> 5;
  const int wm = wave & 1;                 // 2 row sub-tiles of 32
  const int wn = wave >> 1;                // 4 col sub-tiles of 32
  const int half = lane >> 4;
  const int l16  = lane & 15;

  f32x8 acc[2][2];
  for (int i = 0; i < 2; ++i)
    for (int j = 0; j < 2; ++j)
      for (int r = 0; r < 8; ++r) acc[i][j][r] = 0.0f;

  // staging assignments
  const int arow = tid >> 2;               // 0..63
  const int ak0  = (tid & 3) * 8;          // 0,8,16,24
  const int bk0  = (tid >> 4) * 2;         // even k: 0,2,..,30
  const int bn0  = (tid & 15) * 8;         // 0..120

  // ---- loop-invariant staging state (hoisted out of the k-loop) ----
  unsigned short* const aDstLds = &ldsA[arow * LDAP + ak0];
  unsigned short* const bDstLds = &ldsB[(size_t)bn0 * LDAP + bk0];
  const float* const   bSrc    = Wb + (size_t)bk0 * D_OUT + nBase + bn0;

  const float* aSrc1 = nullptr;                  // phase1: gathered x row (or null pad row)
  const unsigned short* aSrc2 = nullptr;         // phase2: contiguous bf16 h row
  if (PHASE == 1) {
    int r = rowIdx[arow];
    if (r >= 0) aSrc1 = xf + (size_t)r * D_IN + ak0;
  } else {
    int ar = posBase + arow; if (ar > B_ROWS - 1) ar = B_ROWS - 1;   // clamp (unused rows)
    aSrc2 = hA + (size_t)ar * D_OUT + ak0;
  }
  const unsigned int laA = (unsigned int)(uintptr_t)aDstLds;  // LDS byte offset (addr[31:0])

  // fragment LDS pointers (loop-invariant)
  const unsigned short* const paBase = &ldsA[(wm * 32 + l16) * LDAP];
  const unsigned short* const pbBase = &ldsB[(wn * 32 + l16) * LDAP + half * 8];

  for (int kb = 0; kb < K; kb += TK) {
    // ---- stage A tile (64 x 32) into LDS as bf16, row-major [row][k] ----
    if (PHASE == 1) {
      float4 f0 = make_float4(0.f, 0.f, 0.f, 0.f), f1 = f0;
      if (aSrc1) {
        const float4* src = reinterpret_cast<const float4*>(aSrc1 + kb);
        f0 = src[0]; f1 = src[1];
      }
      uint4 packed = make_uint4(f2bf2(f0.x, f0.y), f2bf2(f0.z, f0.w),
                                f2bf2(f1.x, f1.y), f2bf2(f1.z, f1.w));
      *reinterpret_cast<uint4*>(aDstLds) = packed;
    } else {
      // CDNA5 async memory->LDS copy (bf16 tile, no conversion needed); ASYNCcnt-tracked
      uint64_t ga = (uint64_t)(uintptr_t)(aSrc2 + kb);
      asm volatile("global_load_async_to_lds_b128 %0, %1, off"
                   :: "v"(laA), "v"(ga) : "memory");
    }
    // ---- stage B tile (32 x 128) transposed into LDS [n][k], f32 -> bf16 ----
    // each thread covers k = bk0,bk0+1 and n = bn0..bn0+7; writes packed dwords along k
    {
      const float* s0 = bSrc + (size_t)kb * D_OUT;
      float4 a0 = *reinterpret_cast<const float4*>(s0);
      float4 a1 = *reinterpret_cast<const float4*>(s0 + 4);
      float4 b0 = *reinterpret_cast<const float4*>(s0 + D_OUT);
      float4 b1 = *reinterpret_cast<const float4*>(s0 + D_OUT + 4);
      unsigned int u0 = f2bf2(a0.x, b0.x), u1 = f2bf2(a0.y, b0.y);
      unsigned int u2 = f2bf2(a0.z, b0.z), u3 = f2bf2(a0.w, b0.w);
      unsigned int u4 = f2bf2(a1.x, b1.x), u5 = f2bf2(a1.y, b1.y);
      unsigned int u6 = f2bf2(a1.z, b1.z), u7 = f2bf2(a1.w, b1.w);
      *reinterpret_cast<unsigned int*>(bDstLds + 0 * LDAP) = u0;
      *reinterpret_cast<unsigned int*>(bDstLds + 1 * LDAP) = u1;
      *reinterpret_cast<unsigned int*>(bDstLds + 2 * LDAP) = u2;
      *reinterpret_cast<unsigned int*>(bDstLds + 3 * LDAP) = u3;
      *reinterpret_cast<unsigned int*>(bDstLds + 4 * LDAP) = u4;
      *reinterpret_cast<unsigned int*>(bDstLds + 5 * LDAP) = u5;
      *reinterpret_cast<unsigned int*>(bDstLds + 6 * LDAP) = u6;
      *reinterpret_cast<unsigned int*>(bDstLds + 7 * LDAP) = u7;
      if (kb + TK < K)                      // speculative prefetch of next weight tile
        __builtin_prefetch(s0 + (size_t)TK * D_OUT, 0, 0);
    }
    if (PHASE == 2) wait_asynccnt0();       // async LDS writes are not covered by DScnt
    __syncthreads();

    // ---- build fragments per ISA VGPR layouts, issue 4 WMMAs ----
    FragAB a[2], b[2];
    for (int i = 0; i < 2; ++i) {
      const unsigned short* pa = paBase + i * 16 * LDAP;
      a[i].q[0] = *reinterpret_cast<const uint4*>(pa + half * 8);        // K: half*8 .. +7
      a[i].q[1] = *reinterpret_cast<const uint4*>(pa + 16 + half * 8);   // K: 16+half*8 .. +7
    }
    for (int j = 0; j < 2; ++j) {
      const unsigned short* pb = pbBase + j * 16 * LDAP;
      b[j].q[0] = *reinterpret_cast<const uint4*>(pb);                   // K: half*8 .. +7
      b[j].q[1] = *reinterpret_cast<const uint4*>(pb + 16);              // K: 16+half*8 .. +7
    }
    for (int i = 0; i < 2; ++i)
      for (int j = 0; j < 2; ++j)
        acc[i][j] = __builtin_amdgcn_wmma_f32_16x16x32_bf16(
            false, a[i].v, false, b[j].v, (short)0, acc[i][j], false, false);
    __syncthreads();
  }

  // ---- epilogue ----
  for (int i = 0; i < 2; ++i)
    for (int j = 0; j < 2; ++j)
      for (int r = 0; r < 8; ++r) {
        int rowInTile = wm * 32 + i * 16 + half * 8 + r;
        int p = posBase + rowInTile;
        if (p < segE) {
          int col = nBase + wn * 32 + j * 16 + l16;
          size_t idx = (size_t)p * D_OUT + col;
          float v = acc[i][j][r];
          if (PHASE == 1) {
            eBuf[idx] = v;                                   // embed1
            float g = v / (1.0f + __expf(-1.702f * v));      // x*sigmoid(1.702x)
            hOut[idx] = f2bf(g);
          } else {
            eBuf[idx] = v + eBuf[idx];                       // s = embed2 + embed1 (in place)
          }
        }
      }
}

// ---------------- LayerNorm + L2 normalize + scatter ----------------

__device__ __forceinline__ float block_reduce(float v, float* red) {
  int tid = threadIdx.x;
  red[tid] = v;
  __syncthreads();
  for (int s = 128; s > 0; s >>= 1) {
    if (tid < s) red[tid] += red[tid + s];
    __syncthreads();
  }
  float r = red[0];
  __syncthreads();
  return r;
}

__global__ __launch_bounds__(256)
void ln_l2_kernel(const float* __restrict__ sBuf, const int* __restrict__ perm,
                  const int* __restrict__ cls, const float* __restrict__ gamma,
                  const float* __restrict__ beta, float* __restrict__ out) {
  __shared__ float red[256];
  const int p = blockIdx.x;
  const int b = perm[p];
  const int e = cls[b];
  const int tid = threadIdx.x;

  float4 sv = *reinterpret_cast<const float4*>(sBuf + (size_t)p * D_OUT + tid * 4);
  float sum = sv.x + sv.y + sv.z + sv.w;
  float ssq = sv.x * sv.x + sv.y * sv.y + sv.z * sv.z + sv.w * sv.w;
  float sumAll = block_reduce(sum, red);
  float ssqAll = block_reduce(ssq, red);
  float mean = sumAll * (1.0f / D_OUT);
  float var  = ssqAll * (1.0f / D_OUT) - mean * mean;
  float rstd = rsqrtf(var + 1e-5f);

  float4 g  = *reinterpret_cast<const float4*>(gamma + (size_t)e * D_OUT + tid * 4);
  float4 bt = *reinterpret_cast<const float4*>(beta  + (size_t)e * D_OUT + tid * 4);
  float y0 = (sv.x - mean) * rstd * g.x + bt.x;
  float y1 = (sv.y - mean) * rstd * g.y + bt.y;
  float y2 = (sv.z - mean) * rstd * g.z + bt.z;
  float y3 = (sv.w - mean) * rstd * g.w + bt.w;

  float n2 = block_reduce(y0 * y0 + y1 * y1 + y2 * y2 + y3 * y3, red);
  float inv = rsqrtf(n2);
  float4 o = make_float4(y0 * inv, y1 * inv, y2 * inv, y3 * inv);
  *reinterpret_cast<float4*>(out + (size_t)b * D_OUT + tid * 4) = o;
}

// ---------------- launch ----------------

static inline size_t align256(size_t v) { return (v + 255) & ~(size_t)255; }

extern "C" void kernel_launch(void* const* d_in, const int* in_sizes, int n_in,
                              void* d_out, int out_size, void* d_ws, size_t ws_size,
                              hipStream_t stream) {
  (void)in_sizes; (void)n_in; (void)out_size; (void)ws_size;
  const float* x      = (const float*)d_in[0];   // [4096,768]
  const float* logits = (const float*)d_in[1];   // [4096,10]
  const float* W1     = (const float*)d_in[2];   // [10,768,1024]
  const float* W2     = (const float*)d_in[3];   // [10,1024,1024]
  const float* gamma  = (const float*)d_in[4];   // [10,1024]
  const float* beta   = (const float*)d_in[5];   // [10,1024]

  float* out       = (float*)d_out;                          // [4096,1024]
  float* outLogits = out + (size_t)B_ROWS * D_OUT;           // [4096,10]

  char* ws = (char*)d_ws;
  int* cls    = (int*)ws;                 ws += align256((size_t)B_ROWS * 4);
  int* counts = (int*)ws;                 ws += align256(N_EXP * 4);
  int* segOff = (int*)ws;                 ws += align256((N_EXP + 1) * 4);
  int* cursor = (int*)ws;                 ws += align256(N_EXP * 4);
  int* perm   = (int*)ws;                 ws += align256((size_t)B_ROWS * 4);
  float* sBuf = (float*)ws;               ws += align256((size_t)B_ROWS * D_OUT * 4);  // embed1 -> s
  unsigned short* hBuf = (unsigned short*)ws;  // bf16 gelu activations [4096,1024]

  init_counts_kernel<<<1, 32, 0, stream>>>(counts, cursor);
  route_kernel<<<B_ROWS / 256, 256, 0, stream>>>(logits, cls, counts, outLogits);
  scan_kernel<<<1, 32, 0, stream>>>(counts, segOff);
  scatter_kernel<<<B_ROWS / 256, 256, 0, stream>>>(cls, segOff, cursor, perm);

  dim3 grid(D_OUT / TN, B_ROWS / TM, N_EXP);   // (8, 64, 10); empty segment tiles exit early
  moe_gemm_kernel<1><<<grid, NTHREADS, 0, stream>>>(x, nullptr, W1, perm, segOff,
                                                    sBuf, hBuf, D_IN);
  moe_gemm_kernel<2><<<grid, NTHREADS, 0, stream>>>(nullptr, hBuf, W2, perm, segOff,
                                                    sBuf, nullptr, D_OUT);
  ln_l2_kernel<<<B_ROWS, 256, 0, stream>>>(sBuf, perm, cls, gamma, beta, out);
}
